// BayesianMaskedLinear_13365938225783
// MI455X (gfx1250) — matrix operand order repaired and checked by reading, not compiled, not run on hardware
//
#include <hip/hip_runtime.h>
#include <hip/hip_bf16.h>
#include <stdint.h>

// ---------------------------------------------------------------------------
// BayesianMaskedLinear on MI455X (gfx1250): fused weight-sampling GEMMs with
// v_wmma_f32_16x16x32_bf16, async global->LDS staging of activations,
// mask folded into K-loop bounds + tile-edge zeroing, bias+ReLU epilogue.
// ---------------------------------------------------------------------------

typedef __attribute__((ext_vector_type(16))) __bf16 v16bf;
typedef __attribute__((ext_vector_type(8)))  __bf16 v8bf;
typedef __attribute__((ext_vector_type(8)))  float  v8f;

union Frag16 { v16bf v; v8bf h[2]; };

#define DTOT   4096
#define MROWS  64
#define NUMMC  5
#define KBLK   64     // K elements staged per iteration
#define NTW    128    // N tile width per workgroup (two 64-feature blocks)

// ---- counter-based RNG mixing ---------------------------------------------
__device__ __forceinline__ uint32_t mix3(uint32_t a, uint32_t b, uint32_t c) {
    uint32_t h = a * 0x9E3779B1u ^ b * 0x85EBCA77u ^ c * 0xC2B2AE3Du;
    h ^= h >> 16; h *= 0x7FEB352Du;
    h ^= h >> 15; h *= 0x846CA68Bu;
    h ^= h >> 16;
    return h;
}
// Branchless softplus: log1p(exp(x)) ~= log(1+exp(min(x,20))) + max(x-20,0)
__device__ __forceinline__ float softplus_f(float x) {
    float xe = fminf(x, 20.0f);
    return __logf(1.0f + __expf(xe)) + fmaxf(x - 20.0f, 0.0f);
}

// ---- prep: fp32 x -> bf16 --------------------------------------------------
__global__ __launch_bounds__(256) void prep_bf16(const float* __restrict__ src,
                                                 __bf16* __restrict__ dst, int n) {
    int i = blockIdx.x * blockDim.x + threadIdx.x;
    if (i < n) dst[i] = (__bf16)src[i];
}

// ---- fused sampled-weight GEMM --------------------------------------------
// STAGE 1: A = x_bf (shared across mc), D = relu(A @ W1^T + b1) -> h_bf[mc]
// STAGE 2: A = h_bf[mc],                D = A @ W2^T + b2       -> outs[mc] (f32)
// grid = (32 N-tiles of 128, 5 mc); block = 512 threads = 16 wave32 waves.
// Each wave owns one 16(M)x16(N) tile pair {noff, noff+64}.
template <int STAGE>
__global__ __launch_bounds__(512)
void sampled_gemm(const float* __restrict__ Wmu, const float* __restrict__ Wrho,
                  const float* __restrict__ bmu, const float* __restrict__ brho,
                  const __bf16* __restrict__ Abase,
                  __bf16* __restrict__ Hout, float* __restrict__ Fout) {
    const int nblk = blockIdx.x;        // 0..31: output features [nblk*128, +128)
    const int mc   = blockIdx.y;        // 0..4
    const int tid  = threadIdx.x;
    const int wave = tid >> 5;          // wave32: 16 waves
    const int lane = tid & 31;
    const int moff = (wave >> 2) * 16;  // 0,16,32,48
    const int noff = (wave & 3) * 16;   // 0..48 (tile pair: noff, noff+64)
    const int nbase = nblk * NTW;

    __shared__ __align__(16) __bf16 lds_x[MROWS * KBLK];  // [m][k]  8 KB
    __shared__ __align__(16) __bf16 lds_w[NTW * KBLK];    // [n][k] 16 KB (= W^T tile)

    const __bf16* __restrict__ A =
        Abase + (STAGE == 2 ? (size_t)mc * MROWS * DTOT : (size_t)0);

    const uint32_t wsalt = 0x51ED2700u ^ (STAGE << 20) ^ mc;
    const uint32_t bsalt = 0xB1A5ED00u ^ (STAGE << 20) ^ mc;

    v8f acc0 = {}, acc1 = {};

    // Block-upper-triangular mask: first n-block of this tile is 2*nblk,
    // so K starts at 2*nblk*64 == nbase. Tile-internal mask handled below.
    for (int kbase = nbase; kbase < DTOT; kbase += KBLK) {
        // --- async-stage activation tile: 64x64 bf16 = 8KB, one b128/thread ---
        {
            int e = tid * 8;                          // bf16 element index
            int m = e >> 6, k = e & 63;
            uint32_t ldsoff = (uint32_t)(uintptr_t)(lds_x + e);
            const __bf16* gp = A + (size_t)m * DTOT + kbase + k;
            asm volatile("global_load_async_to_lds_b128 %0, %1, off"
                         :: "v"(ldsoff), "v"(gp) : "memory");
        }
        // --- generate sampled W^T tile (overlaps the async load) -----------
        // lds_w[n][k] = mask * (mu + z * softplus(rho)); pairs share Box-Muller.
        for (int p = tid; p < (NTW * KBLK) / 2; p += 512) {
            int k2 = (p & 31) << 1;                   // 0,2,...,62
            int n  = p >> 5;                          // 0..127
            int gn = nbase + n, gk = kbase + k2;
            size_t idx = (size_t)gn * DTOT + gk;
            float mu0 = Wmu[idx], mu1 = Wmu[idx + 1];
            float sp0 = softplus_f(Wrho[idx]);
            float sp1 = softplus_f(Wrho[idx + 1]);
            if (kbase + KBLK < DTOT) {                // prefetch next K tile
                __builtin_prefetch(&Wmu[idx + KBLK], 0, 0);
                __builtin_prefetch(&Wrho[idx + KBLK], 0, 0);
            }
            uint32_t u1 = mix3(wsalt, (uint32_t)gn, (uint32_t)gk);
            uint32_t u2 = mix3(wsalt ^ 0xDEADBEEFu, (uint32_t)gk, (uint32_t)gn);
            float f1 = ((float)u1 + 1.0f) * 2.3283064e-10f;  // (0,1]
            float f2 = (float)u2 * 2.3283064e-10f;           // [0,1)
            float sn, cs;
            __sincosf(6.28318530718f * f2, &sn, &cs);
            float r  = __fsqrt_rn(-2.0f * __logf(f1));
            float z0 = r * cs, z1 = r * sn;
            float msk = ((gk >> 6) >= (gn >> 6)) ? 1.0f : 0.0f;
            lds_w[n * KBLK + k2]     = (__bf16)(msk * (mu0 + z0 * sp0));
            lds_w[n * KBLK + k2 + 1] = (__bf16)(msk * (mu1 + z1 * sp1));
        }
        asm volatile("s_wait_asynccnt 0x0" ::: "memory");
        __syncthreads();

        // --- 4 WMMAs per wave: 2 K-halves x 2 N-tiles, A fragment reused ---
        #pragma unroll
        for (int kh = 0; kh < 2; ++kh) {
            const int kb   = ((lane < 16) ? 0 : 16) + kh * 32;
            const int mrow = moff + (lane & 15);
            Frag16 a, b0, b1;
            a.h[0] = *(const v8bf*)(lds_x + mrow * KBLK + kb);
            a.h[1] = *(const v8bf*)(lds_x + mrow * KBLK + kb + 8);
            const int kb2  = ((lane < 16) ? 0 : 16) + kh * 32;
            const int nrow = noff + (lane & 15);
            b0.h[0] = *(const v8bf*)(lds_w + nrow * KBLK + kb2);
            b0.h[1] = *(const v8bf*)(lds_w + nrow * KBLK + kb2 + 8);
            b1.h[0] = *(const v8bf*)(lds_w + (nrow + 64) * KBLK + kb2);
            b1.h[1] = *(const v8bf*)(lds_w + (nrow + 64) * KBLK + kb2 + 8);
            acc0 = __builtin_amdgcn_wmma_f32_16x16x32_bf16(
                false, a.v, false, b0.v, (short)0, acc0, false, false);
            acc1 = __builtin_amdgcn_wmma_f32_16x16x32_bf16(
                false, a.v, false, b1.v, (short)0, acc1, false, false);
        }
        __syncthreads();
    }

    // --- epilogue: sampled bias (+ReLU for stage 1) ------------------------
    // C layout: VGPR r holds M = moff + r + 8*(lane>=16), N = lane%16.
    const int ncol0 = nbase + noff + (lane & 15);
    const int ncol1 = ncol0 + 64;
    uint32_t ub0 = mix3(bsalt, (uint32_t)ncol0, 1u);
    uint32_t ub1 = mix3(bsalt, (uint32_t)ncol1, 1u);
    uint32_t vb0 = mix3(bsalt ^ 0xDEADBEEFu, 1u, (uint32_t)ncol0);
    uint32_t vb1 = mix3(bsalt ^ 0xDEADBEEFu, 1u, (uint32_t)ncol1);
    float z0 = __fsqrt_rn(-2.0f * __logf(((float)ub0 + 1.0f) * 2.3283064e-10f)) *
               __cosf(6.28318530718f * ((float)vb0 * 2.3283064e-10f));
    float z1 = __fsqrt_rn(-2.0f * __logf(((float)ub1 + 1.0f) * 2.3283064e-10f)) *
               __cosf(6.28318530718f * ((float)vb1 * 2.3283064e-10f));
    float bias0 = bmu[ncol0] + z0 * softplus_f(brho[ncol0]);
    float bias1 = bmu[ncol1] + z1 * softplus_f(brho[ncol1]);
    const int mtop = moff + ((lane < 16) ? 0 : 8);
    #pragma unroll
    for (int r = 0; r < 8; ++r) {
        int m = mtop + r;
        float v0 = acc0[r] + bias0;
        float v1 = acc1[r] + bias1;
        if (STAGE == 1) {
            v0 = fmaxf(v0, 0.0f);
            v1 = fmaxf(v1, 0.0f);
            size_t base = (size_t)mc * MROWS * DTOT + (size_t)m * DTOT;
            Hout[base + ncol0] = (__bf16)v0;
            Hout[base + ncol1] = (__bf16)v1;
        } else {
            size_t base = (size_t)mc * MROWS * DTOT + (size_t)m * DTOT;
            Fout[base + ncol0] = v0;
            Fout[base + ncol1] = v1;
        }
    }
}

// ---- mean / std(ddof=1) over 5 MC samples ---------------------------------
__global__ __launch_bounds__(256) void mc_reduce(const float* __restrict__ outs,
                                                 float* __restrict__ out, int n) {
    int p = blockIdx.x * blockDim.x + threadIdx.x;
    if (p >= n) return;
    float v[NUMMC], s = 0.0f;
    #pragma unroll
    for (int i = 0; i < NUMMC; ++i) { v[i] = outs[(size_t)i * n + p]; s += v[i]; }
    float mean = s * (1.0f / NUMMC);
    float ss = 0.0f;
    #pragma unroll
    for (int i = 0; i < NUMMC; ++i) { float d = v[i] - mean; ss += d * d; }
    out[p]     = mean;
    out[n + p] = __fsqrt_rn(ss * (1.0f / (NUMMC - 1)));
}

extern "C" void kernel_launch(void* const* d_in, const int* in_sizes, int n_in,
                              void* d_out, int out_size, void* d_ws, size_t ws_size,
                              hipStream_t stream) {
    const float* x     = (const float*)d_in[0];
    const float* wmu1  = (const float*)d_in[1];
    const float* wrho1 = (const float*)d_in[2];
    const float* bmu1  = (const float*)d_in[3];
    const float* brho1 = (const float*)d_in[4];
    const float* wmu2  = (const float*)d_in[5];
    const float* wrho2 = (const float*)d_in[6];
    const float* bmu2  = (const float*)d_in[7];
    const float* brho2 = (const float*)d_in[8];

    const int NELEM = MROWS * DTOT;                 // 262144
    char* ws = (char*)d_ws;
    __bf16* x_bf = (__bf16*)ws;                                    // 512 KB
    __bf16* h_bf = (__bf16*)(ws + (size_t)NELEM * 2);              // 2.5 MB (5 mc)
    float*  outs = (float*)(ws + (size_t)NELEM * 2 * (1 + NUMMC)); // 5 MB  (5 mc)

    prep_bf16<<<(NELEM + 255) / 256, 256, 0, stream>>>(x, x_bf, NELEM);

    dim3 grid(DTOT / NTW, NUMMC);                   // (32 N-tiles, 5 mc)
    sampled_gemm<1><<<grid, 512, 0, stream>>>(wmu1, wrho1, bmu1, brho1,
                                              x_bf, h_bf, nullptr);
    sampled_gemm<2><<<grid, 512, 0, stream>>>(wmu2, wrho2, bmu2, brho2,
                                              h_bf, nullptr, outs);

    mc_reduce<<<(NELEM + 255) / 256, 256, 0, stream>>>(outs, (float*)d_out, NELEM);
}
